// GraphConvolution_47605417508915
// MI455X (gfx1250) — matrix operand order repaired and checked by reading, not compiled, not run on hardware
//
#include <hip/hip_runtime.h>
#include <hip/hip_bf16.h>

typedef __attribute__((ext_vector_type(16))) _Float16 v16h;
typedef __attribute__((ext_vector_type(8)))  _Float16 v8h;
typedef __attribute__((ext_vector_type(8)))  float    v8f;

#define HIDC 256
#define CFEAT 128
#define ROW_T 2
#define COL_T 4

// ---------------- WMMA GEMM:  C[Mr x 256] = A[Mr x K](f16) @ Bt[256 x K]^T(f16) + bias
// Mr is a multiple of 64. Each wave computes a 32x64 tile: 2 row-tiles x 4 col-tiles.
// 8 WMMAs : 12 B128 loads per K-step; K-loop unrolled 2x (trip counts 4/8/12 are even)
// so two iterations' load clauses and WMMA groups interleave and the second A-fragment
// wait no longer serializes. A re-read factor = N/64 = 4 (A streams from L2; the 128KB
// B weight panel stays WGP$/L2-resident).
__global__ __launch_bounds__(256)
void k_gemm(const _Float16* __restrict__ A, const _Float16* __restrict__ Bt,
            const float* __restrict__ bias, float* __restrict__ C,
            int Mr, int K, int N)
{
  const int lane = threadIdx.x & 31;
  const int wave = threadIdx.x >> 5;
  const int l16  = lane & 15;
  const int h    = lane >> 4;
  const int n0   = blockIdx.y * (COL_T * 16);
  const int rowBase = (blockIdx.x * 8 + wave) * (ROW_T * 16);
  if (rowBase >= Mr) return;            // wave-uniform; active waves keep EXEC all-1s

  v8f acc[ROW_T][COL_T];
#pragma unroll
  for (int ct = 0; ct < COL_T; ++ct) {
    float bv = bias[n0 + ct * 16 + l16];
#pragma unroll
    for (int rt = 0; rt < ROW_T; ++rt)
#pragma unroll
      for (int i = 0; i < 8; ++i) acc[rt][ct][i] = bv;
  }

  const _Float16* ap[ROW_T];
#pragma unroll
  for (int rt = 0; rt < ROW_T; ++rt)
    ap[rt] = A + (size_t)(rowBase + rt * 16 + l16) * K + h * 16;  // A frag: K = h*16 + i
  const _Float16* bp[COL_T];
#pragma unroll
  for (int ct = 0; ct < COL_T; ++ct)
    bp[ct] = Bt + (size_t)(n0 + ct * 16 + l16) * K + h * 16;      // B frag: K = h*16 + i

#pragma unroll 2
  for (int k0 = 0; k0 < K; k0 += 32) {
    v16h bf[COL_T];
#pragma unroll
    for (int ct = 0; ct < COL_T; ++ct)
      bf[ct] = *(const v16h*)(bp[ct] + k0);                        // 32B (2 x b128)
    union Frag { v16h v; v8h p[2]; } af[ROW_T];
#pragma unroll
    for (int rt = 0; rt < ROW_T; ++rt) {
      af[rt].p[0] = *(const v8h*)(ap[rt] + k0);                    // K = h*16 .. h*16+7
      af[rt].p[1] = *(const v8h*)(ap[rt] + k0 + 8);                // K = h*16+8 ..
    }
#pragma unroll
    for (int rt = 0; rt < ROW_T; ++rt)
#pragma unroll
      for (int ct = 0; ct < COL_T; ++ct)
        acc[rt][ct] = __builtin_amdgcn_wmma_f32_16x16x32_f16(
            false, af[rt].v, false, bf[ct], (short)0, acc[rt][ct], false, false);
  }

#pragma unroll
  for (int rt = 0; rt < ROW_T; ++rt)
#pragma unroll
    for (int ct = 0; ct < COL_T; ++ct) {
      const int n = n0 + ct * 16 + l16;
#pragma unroll
      for (int r = 0; r < 8; ++r) {
        int row = rowBase + rt * 16 + r + 8 * h;                   // C/D layout: M = r + 8*(lane/16)
        C[(size_t)row * N + n] = acc[rt][ct][r];
      }
    }
}

// ---------------- weight transpose + f32->f16:  Wt[n*K+k] = W[k*N+n]
__global__ __launch_bounds__(256)
void k_txp(const float* __restrict__ W, _Float16* __restrict__ Wt, int K, int N) {
  int idx = blockIdx.x * blockDim.x + threadIdx.x;
  if (idx >= K * N) return;
  int k = idx / N, n = idx - k * N;
  Wt[(size_t)n * K + k] = (_Float16)W[idx];
}

// ---------------- f32 -> f16 with zero row padding (ld = 256)
__global__ __launch_bounds__(256)
void k_cvt(const float* __restrict__ Y, _Float16* __restrict__ X, int V, int total) {
  int idx = blockIdx.x * blockDim.x + threadIdx.x;
  if (idx >= total) return;
  int row = idx >> 8;
  X[idx] = (row < V) ? (_Float16)Y[idx] : (_Float16)0.f;
}

// ---------------- bilinear sample (matches reference _vert_align semantics)
__device__ __forceinline__ float bilin(const float* __restrict__ feat, int H, int W,
                                       float vx, float vy, int c) {
  float fx = (vx + 1.f) * 0.5f * (float)(W - 1);
  float fy = (vy + 1.f) * 0.5f * (float)(H - 1);
  float x0 = floorf(fx), y0 = floorf(fy);
  float wx = fx - x0, wy = fy - y0;
  int x0i = (int)x0, y0i = (int)y0;
  float acc = 0.f;
  const size_t cs = (size_t)c * H * W;
#pragma unroll
  for (int dy = 0; dy < 2; ++dy)
#pragma unroll
    for (int dx = 0; dx < 2; ++dx) {
      int yi = y0i + dy, xi = x0i + dx;
      bool valid = (yi >= 0) && (yi < H) && (xi >= 0) && (xi < W);
      int yc = yi < 0 ? 0 : (yi > H - 1 ? H - 1 : yi);
      int xc = xi < 0 ? 0 : (xi > W - 1 ? W - 1 : xi);
      float v = feat[cs + (size_t)yc * W + xc];
      float w = (dy ? wy : 1.f - wy) * (dx ? wx : 1.f - wx);
      acc += valid ? v * w : 0.f;
    }
  return acc;
}

// ---------------- stage-1 input: XH[Mr x 128] (f16) = vert_align(conv64, verts0)
__global__ __launch_bounds__(256)
void k_bin1(const float* __restrict__ feat, const float* __restrict__ verts,
            _Float16* __restrict__ X, int V, int total) {
  int idx = blockIdx.x * blockDim.x + threadIdx.x;
  if (idx >= total) return;
  int row = idx >> 7, c = idx & 127;
  X[idx] = (row < V) ? (_Float16)bilin(feat, 56, 56, verts[row * 3], verts[row * 3 + 1], c)
                     : (_Float16)0.f;
}

// ---------------- stage-2/3 input: XH[Mr x 384] = concat(vert_align(feat,VS), subdiv(Yprev))
__global__ __launch_bounds__(256)
void k_bnext(const float* __restrict__ feat, int H, int W,
             const float* __restrict__ VS, const float* __restrict__ Yprev,
             const int* __restrict__ edges, int Vprev, int Vnew, int total,
             _Float16* __restrict__ X) {
  int idx = blockIdx.x * blockDim.x + threadIdx.x;
  if (idx >= total) return;
  int row = idx / 384, c = idx - row * 384;
  _Float16 out;
  if (row >= Vnew) out = (_Float16)0.f;
  else if (c < CFEAT)
    out = (_Float16)bilin(feat, H, W, VS[row * 3], VS[row * 3 + 1], c);
  else {
    int cc = c - CFEAT; float val;
    if (row < Vprev) val = Yprev[(size_t)row * HIDC + cc];
    else {
      int e = row - Vprev;
      int e0 = edges[2 * e], e1 = edges[2 * e + 1];
      val = 0.5f * (Yprev[(size_t)e0 * HIDC + cc] + Yprev[(size_t)e1 * HIDC + cc]);
    }
    out = (_Float16)val;
  }
  X[idx] = out;
}

// ---------------- edge scatter-add (graph aggregation), 4 channels per thread
__global__ __launch_bounds__(256)
void k_scat(float* __restrict__ Y, const float* __restrict__ T,
            const int* __restrict__ edges, int E) {
  int idx = blockIdx.x * blockDim.x + threadIdx.x;
  if (idx >= E * 64) return;
  int e = idx >> 6, c = (idx & 63) << 2;
  int e0 = edges[2 * e], e1 = edges[2 * e + 1];
  const float4 a = *(const float4*)(T + (size_t)e1 * HIDC + c);
  float* p0 = Y + (size_t)e0 * HIDC + c;
  atomicAdd(p0 + 0, a.x); atomicAdd(p0 + 1, a.y); atomicAdd(p0 + 2, a.z); atomicAdd(p0 + 3, a.w);
  const float4 b = *(const float4*)(T + (size_t)e0 * HIDC + c);
  float* p1 = Y + (size_t)e1 * HIDC + c;
  atomicAdd(p1 + 0, b.x); atomicAdd(p1 + 1, b.y); atomicAdd(p1 + 2, b.z); atomicAdd(p1 + 3, b.w);
}

// ---------------- 3-channel output gconv GEMM part (K=256, N=3): both h0 and h1
__global__ __launch_bounds__(256)
void k_coords(const float* __restrict__ Y, const float* __restrict__ wout,
              const float* __restrict__ bout, float* __restrict__ C0,
              float* __restrict__ C1, int V) {
  int idx = blockIdx.x * blockDim.x + threadIdx.x;
  if (idx >= V * 3) return;
  int v = idx / 3, j = idx - v * 3;
  const float* y  = Y + (size_t)v * HIDC;
  const float* w0 = wout;
  const float* w1 = wout + HIDC * 3;
  float a0 = bout[j], a1 = bout[3 + j];
  for (int k = 0; k < HIDC; ++k) {
    float yv = y[k];
    a0 = fmaf(yv, w0[k * 3 + j], a0);
    a1 = fmaf(yv, w1[k * 3 + j], a1);
  }
  C0[idx] = a0; C1[idx] = a1;
}

__global__ __launch_bounds__(256)
void k_scat3(float* __restrict__ C0, const float* __restrict__ C1,
             const int* __restrict__ edges, int E) {
  int idx = blockIdx.x * blockDim.x + threadIdx.x;
  if (idx >= E * 3) return;
  int e = idx / 3, j = idx - e * 3;
  int e0 = edges[2 * e], e1 = edges[2 * e + 1];
  atomicAdd(&C0[(size_t)e0 * 3 + j], C1[(size_t)e1 * 3 + j]);
  atomicAdd(&C0[(size_t)e1 * 3 + j], C1[(size_t)e0 * 3 + j]);
}

// ---------------- subdivide coords: Vout[v] = v<Vprev ? Cin[v] : 0.5*(Cin[e0]+Cin[e1])
__global__ __launch_bounds__(256)
void k_sub3(const float* __restrict__ Cin, const int* __restrict__ edges,
            float* __restrict__ Vout, int Vprev, int Vnew) {
  int idx = blockIdx.x * blockDim.x + threadIdx.x;
  if (idx >= Vnew * 3) return;
  int v = idx / 3, j = idx - v * 3;
  float val;
  if (v < Vprev) val = Cin[(size_t)v * 3 + j];
  else {
    int e = v - Vprev;
    int e0 = edges[2 * e], e1 = edges[2 * e + 1];
    val = 0.5f * (Cin[(size_t)e0 * 3 + j] + Cin[(size_t)e1 * 3 + j]);
  }
  Vout[idx] = val;
}

static inline int r64i(int v) { return (v + 63) & ~63; }

#define LAUNCH(kern, total, ...) \
  kern<<<dim3((unsigned)(((total) + 255) / 256)), dim3(256), 0, stream>>>(__VA_ARGS__)

extern "C" void kernel_launch(void* const* d_in, const int* in_sizes, int n_in,
                              void* d_out, int out_size, void* d_ws, size_t ws_size,
                              hipStream_t stream) {
  (void)n_in; (void)out_size; (void)ws_size;
  const float* conv[3]  = {(const float*)d_in[0], (const float*)d_in[1], (const float*)d_in[2]};
  const float* verts0   = (const float*)d_in[3];
  const int*   edges[3] = {(const int*)d_in[4], (const int*)d_in[5], (const int*)d_in[6]};
  const float* w_inA[3] = {(const float*)d_in[7],  (const float*)d_in[13], (const float*)d_in[19]};
  const float* b_inA[3] = {(const float*)d_in[8],  (const float*)d_in[14], (const float*)d_in[20]};
  const float* w_hA[3]  = {(const float*)d_in[9],  (const float*)d_in[15], (const float*)d_in[21]};
  const float* b_hA[3]  = {(const float*)d_in[10], (const float*)d_in[16], (const float*)d_in[22]};
  const float* w_outA[3]= {(const float*)d_in[11], (const float*)d_in[17], (const float*)d_in[23]};
  const float* b_outA[3]= {(const float*)d_in[12], (const float*)d_in[18], (const float*)d_in[24]};

  const int V1 = in_sizes[3] / 3;
  const int E[3] = {in_sizes[4] / 2, in_sizes[5] / 2, in_sizes[6] / 2};
  const int V[3] = {V1, V1 + E[0], V1 + E[0] + E[1]};
  const int din[3] = {CFEAT, CFEAT + HIDC, CFEAT + HIDC};
  const int HW[3] = {56, 28, 14};
  const int M3 = r64i(V[2]);

  // ---- workspace carve (256B aligned)
  char* p = (char*)d_ws;
  auto carve = [&](size_t bytes) { void* r = (void*)p; p += (bytes + 255) & ~(size_t)255; return r; };
  _Float16* XH   = (_Float16*)carve((size_t)M3 * 384 * 2);      // f16 GEMM input, padded rows
  float*    Y    = (float*)   carve((size_t)M3 * 256 * 4);      // gconv output / hidden state
  float*    T    = (float*)   carve((size_t)M3 * 256 * 4);      // h1 buffer
  _Float16* WTin = (_Float16*)carve((size_t)2 * 256 * 384 * 2); // transposed f16 input weights
  _Float16* WTh  = (_Float16*)carve((size_t)24 * 256 * 256 * 2);// transposed f16 hidden weights
  float*    C0s  = (float*)   carve((size_t)V[1] * 3 * 4);      // coords (stage 1,2)
  float*    C1   = (float*)   carve((size_t)V[2] * 3 * 4);
  float*    VS   = (float*)   carve((size_t)V[2] * 3 * 4);      // subdivided coords

  auto gemm = [&](const _Float16* A, const _Float16* Bt, const float* bias, float* Cc,
                  int Mr, int K) {
    const int rowsPerBlock = 8 * ROW_T * 16;   // 8 waves x 32 rows = 256
    dim3 grid((unsigned)((Mr + rowsPerBlock - 1) / rowsPerBlock), 256 / (COL_T * 16));
    k_gemm<<<grid, dim3(256), 0, stream>>>(A, Bt, bias, Cc, Mr, K, 256);
  };

  float* prevC0 = nullptr;
  for (int s = 0; s < 3; ++s) {
    const int Vs = V[s], Es = E[s], K = din[s], Mr = r64i(Vs);

    // transpose+convert this stage's weights to f16
    LAUNCH(k_txp, K * 256, w_inA[s], WTin, K, 256);
    LAUNCH(k_txp, K * 256, w_inA[s] + (size_t)K * 256, WTin + (size_t)256 * K, K, 256);
    for (int l = 0; l < 24; ++l)
      LAUNCH(k_txp, 65536, w_hA[s] + (size_t)l * 65536, WTh + (size_t)l * 65536, 256, 256);

    // build f16 stage input
    if (s == 0) {
      LAUNCH(k_bin1, Mr * CFEAT, conv[0], verts0, XH, Vs, Mr * CFEAT);
    } else {
      LAUNCH(k_sub3, Vs * 3, prevC0, edges[s - 1], VS, V[s - 1], Vs);
      LAUNCH(k_bnext, Mr * 384, conv[s], HW[s], HW[s], VS, Y, edges[s - 1],
             V[s - 1], Vs, Mr * 384, XH);
    }

    // input gconv
    gemm(XH, WTin, b_inA[s], Y, Mr, K);
    gemm(XH, WTin + (size_t)256 * K, b_inA[s] + 256, T, Mr, K);
    LAUNCH(k_scat, Es * 64, Y, T, edges[s], Es);

    // 12 hidden gconvs
    for (int l = 0; l < 12; ++l) {
      LAUNCH(k_cvt, Mr * 256, Y, XH, Vs, Mr * 256);
      gemm(XH, WTh + (size_t)(2 * l) * 65536,     b_hA[s] + (size_t)l * 512,       Y, Mr, 256);
      gemm(XH, WTh + (size_t)(2 * l + 1) * 65536, b_hA[s] + (size_t)l * 512 + 256, T, Mr, 256);
      LAUNCH(k_scat, Es * 64, Y, T, edges[s], Es);
    }

    // output gconv (coords); stage 3 writes directly into d_out
    float* C0 = (s == 2) ? (float*)d_out : C0s;
    LAUNCH(k_coords, Vs * 3, Y, w_outA[s], b_outA[s], C0, C1, Vs);
    LAUNCH(k_scat3, Es * 3, C0, C1, edges[s], Es);
    prevC0 = C0;
  }
}
